// AttendFeedForward_38414187495794
// MI455X (gfx1250) — matrix operand order repaired and checked by reading, not compiled, not run on hardware
//
#include <hip/hip_runtime.h>

// Problem dims
#define B_   32
#define L_   1024
#define INP_ 600
#define HID_ 200
// Padded dims
#define KP1  608           // INP padded to mult of 32 (K of layer1)
#define SNP  640           // s1/s2 packed row stride (bf16), zeros in [600,640)
#define HP   224           // HID padded to mult of 32
#define WNP  256           // weight buffer N extent (covers 2 BN blocks, zeros beyond 224)
// GEMM tiling: block = 128x128, 8 waves (4M x 2N), wave tile 32x64 -> 8 WMMA/step
#define BM   128
#define BN   128
#define BK   32
#define LDSS (BK + 8)      // LDS row stride (bf16 elems) -> 80B, rows 16B aligned

typedef __attribute__((ext_vector_type(16))) __bf16 v16bf;
typedef __attribute__((ext_vector_type(8)))  __bf16 v8bf;
typedef __attribute__((ext_vector_type(8)))  float  v8f;

__device__ __forceinline__ __bf16 f2bf(float f) { return (__bf16)f; }  // v_cvt_pk_bf16_f32 path

__device__ __forceinline__ v16bf frag_pair(const __bf16* p0, const __bf16* p1) {
  v8bf lo = *(const v8bf*)p0;
  v8bf hi = *(const v8bf*)p1;
  v16bf r;
#pragma unroll
  for (int j = 0; j < 8; ++j) { r[j] = lo[j]; r[8 + j] = hi[j]; }
  return r;
}

// A fragment (16x32 bf16): lane = g*16+r holds row r, K = {g*8..g*8+7} U {16+g*8..23+g*8}
__device__ __forceinline__ v16bf frag_a(const __bf16* As, int row, int g) {
  const __bf16* p = As + row * LDSS + g * 8;
  return frag_pair(p, p + 16);
}
// B fragment (32x16 bf16), Bs stored transposed [n][k]: lane col r, K = {g*8..g*8+7} U {16+g*8..23+g*8}
__device__ __forceinline__ v16bf frag_b(const __bf16* Bs, int col, int g) {
  const __bf16* p = Bs + col * LDSS + g * 8;
  return frag_pair(p, p + 16);
}

__device__ __forceinline__ void mma_step(const __bf16* As, const __bf16* Bs,
                                         int wm, int wn, int lane, v8f acc[2][4]) {
  const int g = lane >> 4, r = lane & 15;
  v16bf a[2], b[4];
  a[0] = frag_a(As, wm + r,      g);
  a[1] = frag_a(As, wm + 16 + r, g);
#pragma unroll
  for (int j = 0; j < 4; ++j) b[j] = frag_b(Bs, wn + j * 16 + r, g);
#pragma unroll
  for (int mi = 0; mi < 2; ++mi)
#pragma unroll
    for (int nj = 0; nj < 4; ++nj)
      acc[mi][nj] = __builtin_amdgcn_wmma_f32_16x16x32_bf16(
          false, a[mi], false, b[nj], (short)0, acc[mi][nj], false, false);
}

#define ZERO_ACC(acc) \
  _Pragma("unroll") for (int mi_ = 0; mi_ < 2; ++mi_) \
  _Pragma("unroll") for (int ni_ = 0; ni_ < 4; ++ni_) \
  _Pragma("unroll") for (int e_ = 0; e_ < 8; ++e_) acc[mi_][ni_][e_] = 0.f;

// ---------------- Pack s1/s2: f32 (ld 600) -> zero-padded bf16 (ld 640) ----------------
__global__ __launch_bounds__(256) void k_pack_s(const float* __restrict__ s1,
                                                const float* __restrict__ s2,
                                                __bf16* __restrict__ s1b,
                                                __bf16* __restrict__ s2b) {
  const float* src = blockIdx.y ? s2 : s1;
  __bf16* dst = blockIdx.y ? s2b : s1b;
  int gid = blockIdx.x * 256 + threadIdx.x;       // 32768 rows * 40 segs of 16
  int row = gid / 40, seg = gid % 40;
  const float* s = src + (size_t)row * INP_;
  __bf16* d = dst + (size_t)row * SNP + seg * 16;
#pragma unroll
  for (int i = 0; i < 16; ++i) {
    int c = seg * 16 + i;
    d[i] = f2bf(c < INP_ ? s[c] : 0.f);
  }
}

// ------- Weight packing: f32 row-major -> TRANSPOSED zero-padded bf16 [n][k] -------
__global__ __launch_bounds__(256) void k_pack_w(const float* __restrict__ W1,
                                                const float* __restrict__ W2,
                                                __bf16* __restrict__ w1t,
                                                __bf16* __restrict__ w2t) {
  int i = blockIdx.x * 256 + threadIdx.x;
  const int n1 = WNP * KP1;                       // w1t: 256 n-rows x 608 k-cols
  if (i < n1) {
    int n = i / KP1, k = i % KP1;
    w1t[i] = f2bf((n < HID_ && k < INP_) ? W1[k * HID_ + n] : 0.f);
  } else if (i < n1 + WNP * HP) {                 // w2t: 256 n-rows x 224 k-cols
    int j = i - n1, n = j / HP, k = j % HP;
    w2t[j] = f2bf((n < HID_ && k < HID_) ? W2[k * HID_ + n] : 0.f);
  }
}

// ---------------- Layer 1: T = relu(Xb @ W1 + b1), Xb bf16 (ld SNP), T bf16 (ld HP) ----
__global__ __launch_bounds__(256) void k_gemm1(const __bf16* __restrict__ s1b,
                                               const __bf16* __restrict__ s2b,
                                               const __bf16* __restrict__ w1t,
                                               const float* __restrict__ b1,
                                               __bf16* __restrict__ t1,
                                               __bf16* __restrict__ t2) {
  __shared__ alignas(16) __bf16 As[BM * LDSS];
  __shared__ alignas(16) __bf16 Bs[BN * LDSS];
  const int tid = threadIdx.x, lane = tid & 31, wave = tid >> 5;
  const int wm = (wave >> 1) * 32, wn = (wave & 1) * 64;
  const int n0 = blockIdx.x * BN, m0 = blockIdx.y * BM;
  const __bf16* X = blockIdx.z ? s2b : s1b;
  __bf16* T = blockIdx.z ? t2 : t1;
  v8f acc[2][4]; ZERO_ACC(acc);
  const int ar = tid >> 1, ac = (tid & 1) * 16;
  const __bf16* asrc = X + (size_t)(m0 + ar) * SNP;
  const __bf16* bsrc = w1t + (size_t)(n0 + (tid >> 1)) * KP1;  // B: n-row, contiguous k
  for (int kk = 0; kk < KP1; kk += BK) {
    if (kk + BK < KP1) __builtin_prefetch(asrc + kk + BK + ac, 0, 1);
#pragma unroll
    for (int i = 0; i < 16; ++i) As[ar * LDSS + ac + i] = asrc[kk + ac + i];
#pragma unroll
    for (int i = 0; i < 16; ++i) Bs[ar * LDSS + ac + i] = bsrc[kk + ac + i];
    __syncthreads();
    mma_step(As, Bs, wm, wn, lane, acc);
    __syncthreads();
  }
  const int g = lane >> 4, cN = lane & 15;
#pragma unroll
  for (int mi = 0; mi < 2; ++mi)
#pragma unroll
    for (int nj = 0; nj < 4; ++nj) {
      int n = n0 + wn + nj * 16 + cN;
      if (n < HP) {
        float bias = (n < HID_) ? b1[n] : 0.f;
#pragma unroll
        for (int r8 = 0; r8 < 8; ++r8) {
          int row = m0 + wm + mi * 16 + g * 8 + r8;
          T[(size_t)row * HP + n] = f2bf(fmaxf(acc[mi][nj][r8] + bias, 0.f));
        }
      }
    }
}

// ---------------- Layer 2: H = relu(T @ W2 + b2), all bf16 in, bf16 out ----------------
__global__ __launch_bounds__(256) void k_gemm2(const __bf16* __restrict__ t1,
                                               const __bf16* __restrict__ t2,
                                               const __bf16* __restrict__ w2t,
                                               const float* __restrict__ b2,
                                               __bf16* __restrict__ h1,
                                               __bf16* __restrict__ h2) {
  __shared__ alignas(16) __bf16 As[BM * LDSS];
  __shared__ alignas(16) __bf16 Bs[BN * LDSS];
  const int tid = threadIdx.x, lane = tid & 31, wave = tid >> 5;
  const int wm = (wave >> 1) * 32, wn = (wave & 1) * 64;
  const int n0 = blockIdx.x * BN, m0 = blockIdx.y * BM;
  const __bf16* T = blockIdx.z ? t2 : t1;
  __bf16* H = blockIdx.z ? h2 : h1;
  v8f acc[2][4]; ZERO_ACC(acc);
  const int ar = tid >> 1, ac = (tid & 1) * 16;
  const __bf16* asrc = T + (size_t)(m0 + ar) * HP;
  const __bf16* bsrc = w2t + (size_t)(n0 + (tid >> 1)) * HP;   // B: n-row, contiguous k
  for (int kk = 0; kk < HP; kk += BK) {
    if (kk + BK < HP) __builtin_prefetch(asrc + kk + BK + ac, 0, 1);
#pragma unroll
    for (int i = 0; i < 16; ++i) As[ar * LDSS + ac + i] = asrc[kk + ac + i];
#pragma unroll
    for (int i = 0; i < 16; ++i) Bs[ar * LDSS + ac + i] = bsrc[kk + ac + i];
    __syncthreads();
    mma_step(As, Bs, wm, wn, lane, acc);
    __syncthreads();
  }
  const int g = lane >> 4, cN = lane & 15;
#pragma unroll
  for (int mi = 0; mi < 2; ++mi)
#pragma unroll
    for (int nj = 0; nj < 4; ++nj) {
      int n = n0 + wn + nj * 16 + cN;
      if (n < HP) {
        float bias = (n < HID_) ? b2[n] : 0.f;
#pragma unroll
        for (int r8 = 0; r8 < 8; ++r8) {
          int row = m0 + wm + mi * 16 + g * 8 + r8;
          H[(size_t)row * HP + n] = f2bf(fmaxf(acc[mi][nj][r8] + bias, 0.f));
        }
      }
    }
}

// ---------------- e[b,l,m] = sum_d h1[b,l,d] * h2[b,m,d], f32 out ----------------
__global__ __launch_bounds__(256) void k_gemm_e(const __bf16* __restrict__ h1,
                                                const __bf16* __restrict__ h2,
                                                float* __restrict__ E) {
  __shared__ alignas(16) __bf16 As[BM * LDSS];
  __shared__ alignas(16) __bf16 Bs[BN * LDSS];
  const int tid = threadIdx.x, lane = tid & 31, wave = tid >> 5;
  const int wm = (wave >> 1) * 32, wn = (wave & 1) * 64;
  const int n0 = blockIdx.x * BN, m0 = blockIdx.y * BM, b = blockIdx.z;
  const __bf16* A = h1 + (size_t)b * L_ * HP;
  const __bf16* Bm = h2 + (size_t)b * L_ * HP;
  float* Eb = E + ((size_t)b << 20);
  v8f acc[2][4]; ZERO_ACC(acc);
  const int ar = tid >> 1, ac = (tid & 1) * 16;
  const __bf16* asrc = A + (size_t)(m0 + ar) * HP;
  const __bf16* bsrc = Bm + (size_t)(n0 + ar) * HP;
  for (int kk = 0; kk < HP; kk += BK) {
    if (kk + BK < HP) {
      __builtin_prefetch(asrc + kk + BK + ac, 0, 1);
      __builtin_prefetch(bsrc + kk + BK + ac, 0, 1);
    }
#pragma unroll
    for (int i = 0; i < 16; ++i) As[ar * LDSS + ac + i] = asrc[kk + ac + i];
#pragma unroll
    for (int i = 0; i < 16; ++i) Bs[ar * LDSS + ac + i] = bsrc[kk + ac + i];
    __syncthreads();
    mma_step(As, Bs, wm, wn, lane, acc);
    __syncthreads();
  }
  const int g = lane >> 4, cN = lane & 15;
#pragma unroll
  for (int mi = 0; mi < 2; ++mi)
#pragma unroll
    for (int nj = 0; nj < 4; ++nj)
#pragma unroll
      for (int r8 = 0; r8 < 8; ++r8) {
        int row = m0 + wm + mi * 16 + g * 8 + r8;
        int col = n0 + wn + nj * 16 + cN;
        Eb[(size_t)row * L_ + col] = acc[mi][nj][r8];
      }
}

// ---------------- Online softmax stats: per-column (reduce over l) ----------------
__global__ __launch_bounds__(256) void k_colstats(const float* __restrict__ E,
                                                  const float* __restrict__ mask1,
                                                  float* __restrict__ cmax,
                                                  float* __restrict__ csum) {
  const int b = blockIdx.y;
  const int m = blockIdx.x * 256 + threadIdx.x;
  const float* Eb = E + ((size_t)b << 20);
  float mx = -3.0e38f, s = 0.f;
  for (int l = 0; l < L_; ++l) {
    float x = Eb[(size_t)l * L_ + m] * mask1[b * L_ + l];
    float nm = fmaxf(mx, x);
    s = s * __expf(mx - nm) + __expf(x - nm);
    mx = nm;
  }
  cmax[b * L_ + m] = mx;
  csum[b * L_ + m] = s;
}

// ---------------- Online softmax stats: per-row (reduce over m) ----------------
__global__ __launch_bounds__(256) void k_rowstats(const float* __restrict__ E,
                                                  const float* __restrict__ mask2,
                                                  float* __restrict__ rmax,
                                                  float* __restrict__ rsum) {
  const int b = blockIdx.y, l = blockIdx.x, t = threadIdx.x;
  const float* row = E + ((size_t)b << 20) + (size_t)l * L_;
  float mx = -3.0e38f, s = 0.f;
  for (int c = t; c < L_; c += 256) {
    float x = row[c] * mask2[b * L_ + c];
    float nm = fmaxf(mx, x);
    s = s * __expf(mx - nm) + __expf(x - nm);
    mx = nm;
  }
  __shared__ float sm[256], ss[256];
  sm[t] = mx; ss[t] = s;
  __syncthreads();
  for (int off = 128; off > 0; off >>= 1) {
    if (t < off) {
      float m2 = sm[t + off], s2 = ss[t + off];
      float nm = fmaxf(sm[t], m2);
      ss[t] = ss[t] * __expf(sm[t] - nm) + s2 * __expf(m2 - nm);
      sm[t] = nm;
    }
    __syncthreads();
  }
  if (t == 0) { rmax[b * L_ + l] = sm[0]; rsum[b * L_ + l] = ss[0]; }
}

// s-tile staging for alphas/betas: 4x4 register transpose, b64 in / b64 out
__device__ __forceinline__ void stage_s_tile(__bf16* Bs, const __bf16* sbase,
                                             int l4, int d4) {
  __bf16 rr[4][4];
#pragma unroll
  for (int j = 0; j < 4; ++j) {
    const __bf16* p = sbase + (size_t)j * SNP;
#pragma unroll
    for (int i = 0; i < 4; ++i) rr[j][i] = p[i];
  }
#pragma unroll
  for (int i = 0; i < 4; ++i) {
    __bf16* q = Bs + (d4 + i) * LDSS + l4;
    q[0] = rr[0][i]; q[1] = rr[1][i]; q[2] = rr[2][i]; q[3] = rr[3][i];
  }
}

// -------- alphas[b,m,d] = (mask2[m]/csum[m]) * sum_l exp(e[l,m]*mask1[l]-cmax[m]) * s1[b,l,d]
__global__ __launch_bounds__(256) void k_alphas(const float* __restrict__ E,
                                                const __bf16* __restrict__ s1b,
                                                const float* __restrict__ mask1,
                                                const float* __restrict__ mask2,
                                                const float* __restrict__ cmax,
                                                const float* __restrict__ csum,
                                                float* __restrict__ out) {
  __shared__ alignas(16) __bf16 As[BM * LDSS];
  __shared__ alignas(16) __bf16 Bs[BN * LDSS];
  const int tid = threadIdx.x, lane = tid & 31, wave = tid >> 5;
  const int wm = (wave >> 1) * 32, wn = (wave & 1) * 64;
  const int n0 = blockIdx.x * BN, m0 = blockIdx.y * BM, b = blockIdx.z;
  const float* Eb = E + ((size_t)b << 20);
  v8f acc[2][4]; ZERO_ACC(acc);
  const int ar = tid >> 1, ac = (tid & 1) * 16;   // A: rows = m, cols = l (E transposed)
  const int l4 = (tid & 7) * 4, d4 = (tid >> 3) * 4;  // B: 4l x 4d micro-tile
  const float cm = cmax[b * L_ + m0 + ar];
  for (int kk = 0; kk < L_; kk += BK) {
#pragma unroll
    for (int i = 0; i < 16; ++i) {
      int l = kk + ac + i;
      float e = Eb[(size_t)l * L_ + (m0 + ar)];
      As[ar * LDSS + ac + i] = f2bf(__expf(e * mask1[b * L_ + l] - cm));
    }
    const __bf16* sbase = s1b + (size_t)(b * L_ + kk + l4) * SNP + n0 + d4;
    if (kk + BK < L_) __builtin_prefetch(sbase + (size_t)BK * SNP, 0, 1);
    stage_s_tile(Bs, sbase, l4, d4);
    __syncthreads();
    mma_step(As, Bs, wm, wn, lane, acc);
    __syncthreads();
  }
  const int g = lane >> 4, cN = lane & 15;
#pragma unroll
  for (int mi = 0; mi < 2; ++mi)
#pragma unroll
    for (int r8 = 0; r8 < 8; ++r8) {
      int row = m0 + wm + mi * 16 + g * 8 + r8;
      float scale = mask2[b * L_ + row] / csum[b * L_ + row];
#pragma unroll
      for (int nj = 0; nj < 4; ++nj) {
        int d = n0 + wn + nj * 16 + cN;
        if (d < INP_)
          out[(size_t)(b * L_ + row) * INP_ + d] = acc[mi][nj][r8] * scale;
      }
    }
}

// -------- betas[b,l,d] = (mask1[l]/rsum[l]) * sum_m exp(e[l,m]*mask2[m]-rmax[l]) * s2[b,m,d]
__global__ __launch_bounds__(256) void k_betas(const float* __restrict__ E,
                                               const __bf16* __restrict__ s2b,
                                               const float* __restrict__ mask1,
                                               const float* __restrict__ mask2,
                                               const float* __restrict__ rmax,
                                               const float* __restrict__ rsum,
                                               float* __restrict__ out) {
  __shared__ alignas(16) __bf16 As[BM * LDSS];
  __shared__ alignas(16) __bf16 Bs[BN * LDSS];
  const int tid = threadIdx.x, lane = tid & 31, wave = tid >> 5;
  const int wm = (wave >> 1) * 32, wn = (wave & 1) * 64;
  const int n0 = blockIdx.x * BN, m0 = blockIdx.y * BM, b = blockIdx.z;
  const float* Eb = E + ((size_t)b << 20);
  v8f acc[2][4]; ZERO_ACC(acc);
  const int ar = tid >> 1, ac = (tid & 1) * 16;   // A: rows = l, cols = m (E natural)
  const int l4 = (tid & 7) * 4, d4 = (tid >> 3) * 4;  // B: 4m x 4d micro-tile
  const float rm = rmax[b * L_ + m0 + ar];
  const float* erow = Eb + (size_t)(m0 + ar) * L_;
  for (int kk = 0; kk < L_; kk += BK) {
    if (kk + BK < L_) __builtin_prefetch(erow + kk + BK + ac, 0, 1);
#pragma unroll
    for (int i = 0; i < 16; ++i) {
      int m = kk + ac + i;
      As[ar * LDSS + ac + i] = f2bf(__expf(erow[m] * mask2[b * L_ + m] - rm));
    }
    const __bf16* sbase = s2b + (size_t)(b * L_ + kk + l4) * SNP + n0 + d4;
    stage_s_tile(Bs, sbase, l4, d4);
    __syncthreads();
    mma_step(As, Bs, wm, wn, lane, acc);
    __syncthreads();
  }
  const int g = lane >> 4, cN = lane & 15;
#pragma unroll
  for (int mi = 0; mi < 2; ++mi)
#pragma unroll
    for (int r8 = 0; r8 < 8; ++r8) {
      int row = m0 + wm + mi * 16 + g * 8 + r8;
      float scale = mask1[b * L_ + row] / rsum[b * L_ + row];
#pragma unroll
      for (int nj = 0; nj < 4; ++nj) {
        int d = n0 + wn + nj * 16 + cN;
        if (d < INP_)
          out[(size_t)(b * L_ + row) * INP_ + d] = acc[mi][nj][r8] * scale;
      }
    }
}

extern "C" void kernel_launch(void* const* d_in, const int* in_sizes, int n_in,
                              void* d_out, int out_size, void* d_ws, size_t ws_size,
                              hipStream_t stream) {
  (void)in_sizes; (void)n_in; (void)out_size; (void)ws_size;
  const float* s1    = (const float*)d_in[0];
  const float* s2    = (const float*)d_in[1];
  const float* mask1 = (const float*)d_in[2];
  const float* mask2 = (const float*)d_in[3];
  const float* W1    = (const float*)d_in[4];
  const float* b1    = (const float*)d_in[5];
  const float* W2    = (const float*)d_in[6];
  const float* b2    = (const float*)d_in[7];
  float* out = (float*)d_out;

  char* ws = (char*)d_ws;
  size_t off = 0;
  auto alloc = [&](size_t bytes) -> void* {
    void* p = ws + off;
    off += (bytes + 255) & ~(size_t)255;
    return p;
  };
  const size_t ML = (size_t)B_ * L_;                 // 32768 rows
  __bf16* w1t = (__bf16*)alloc((size_t)WNP * KP1 * 2);
  __bf16* w2t = (__bf16*)alloc((size_t)WNP * HP * 2);
  __bf16* s1b = (__bf16*)alloc(ML * SNP * 2);
  __bf16* s2b = (__bf16*)alloc(ML * SNP * 2);
  __bf16* t1  = (__bf16*)alloc(ML * HP * 2);
  __bf16* t2  = (__bf16*)alloc(ML * HP * 2);
  __bf16* h1  = (__bf16*)alloc(ML * HP * 2);
  __bf16* h2  = (__bf16*)alloc(ML * HP * 2);
  float*  E   = (float*)alloc((size_t)B_ * L_ * L_ * 4);
  float*  cmax = (float*)alloc(ML * 4);
  float*  csum = (float*)alloc(ML * 4);
  float*  rmax = (float*)alloc(ML * 4);
  float*  rsum = (float*)alloc(ML * 4);

  // 0. pack inputs and weights to padded bf16
  k_pack_s<<<dim3((unsigned)(ML * (SNP / 16) / 256), 2), dim3(256), 0, stream>>>(s1, s2, s1b, s2b);
  {
    int total = WNP * KP1 + WNP * HP;
    k_pack_w<<<dim3((total + 255) / 256), dim3(256), 0, stream>>>(W1, W2, w1t, w2t);
  }
  // 1. layer1 for both streams: M=32768, N=224 (2 blocks), K=608
  k_gemm1<<<dim3((HP + BN - 1) / BN, ML / BM, 2), dim3(256), 0, stream>>>(s1b, s2b, w1t, b1, t1, t2);
  // 2. layer2: K=224
  k_gemm2<<<dim3((HP + BN - 1) / BN, ML / BM, 2), dim3(256), 0, stream>>>(t1, t2, w2t, b2, h1, h2);
  // 3. e = h1 @ h2^T per batch: M=N=1024, K=224
  k_gemm_e<<<dim3(L_ / BN, L_ / BM, B_), dim3(256), 0, stream>>>(h1, h2, E);
  // 4. softmax stats
  k_colstats<<<dim3(L_ / 256, B_), dim3(256), 0, stream>>>(E, mask1, cmax, csum);
  k_rowstats<<<dim3(L_, B_), dim3(256), 0, stream>>>(E, mask2, rmax, rsum);
  // 5. alphas / betas GEMMs: M=1024, N=600 (5 blocks), K=1024
  float* alphas = out;
  float* betas  = out + (size_t)B_ * L_ * INP_;
  k_alphas<<<dim3((INP_ + BN - 1) / BN, L_ / BM, B_), dim3(256), 0, stream>>>(
      E, s1b, mask1, mask2, cmax, csum, alphas);
  k_betas<<<dim3((INP_ + BN - 1) / BN, L_ / BM, B_), dim3(256), 0, stream>>>(
      E, s2b, mask1, mask2, rmax, rsum, betas);
}
